// CrossAttention_80745385164878
// MI455X (gfx1250) — matrix operand order repaired and checked by reading, not compiled, not run on hardware
//
#include <hip/hip_runtime.h>

// Problem constants (from reference)
#define BB   4
#define SS   2048
#define EE   1024
#define QDQ  1024
#define KDK  768
#define VDV  768
#define HH   16
#define DD   64

typedef __attribute__((ext_vector_type(16))) __bf16 v16bf;
typedef __attribute__((ext_vector_type(8)))  float  v8f;

// ---------------- helpers ----------------
// Native f32 -> bf16 (lets the backend use gfx1250 packed cvt instead of bit ops)
__device__ __forceinline__ __bf16 f2bf(float x) { return (__bf16)x; }

union Frag16 { float4 f[2]; v16bf v; };

// A fragment (16x32 bf16, row-contiguous source):
// lane half h holds K = {8h..8h+7, 16+8h..16+8h+7}  (ISA 7.12.2, 16-bit A 16x32)
__device__ __forceinline__ v16bf load_a_bf16(const __bf16* row, int half) {
  Frag16 u;
  u.f[0] = *(const float4*)(row + 8 * half);
  u.f[1] = *(const float4*)(row + 16 + 8 * half);
  return u.v;
}
__device__ __forceinline__ v16bf load_a_f32(const float* row, int half) {
  const float* p0 = row + 8 * half;
  const float* p1 = row + 16 + 8 * half;
  float4 a = *(const float4*)p0;
  float4 b = *(const float4*)(p0 + 4);
  float4 c = *(const float4*)p1;
  float4 d = *(const float4*)(p1 + 4);
  v16bf r;
  r[0] = f2bf(a.x);  r[1] = f2bf(a.y);  r[2] = f2bf(a.z);  r[3] = f2bf(a.w);
  r[4] = f2bf(b.x);  r[5] = f2bf(b.y);  r[6] = f2bf(b.z);  r[7] = f2bf(b.w);
  r[8] = f2bf(c.x);  r[9] = f2bf(c.y);  r[10] = f2bf(c.z); r[11] = f2bf(c.w);
  r[12] = f2bf(d.x); r[13] = f2bf(d.y); r[14] = f2bf(d.z); r[15] = f2bf(d.w);
  return r;
}
// B fragment (32x16 bf16, B stored row-contiguous as Bt[N][K]):
// lane half h holds K = {8h..8h+7, 16+8h..16+8h+7}
__device__ __forceinline__ v16bf load_b_bf16(const __bf16* row, int half) {
  Frag16 u;
  u.f[0] = *(const float4*)(row + 8 * half);
  u.f[1] = *(const float4*)(row + 16 + 8 * half);
  return u.v;
}

// ---------------- generic WMMA GEMM ----------------
// C[z] (M x N) = scale * (A[z] (M x K) * Bt[z]^T (K x N)) + bias
// One wave computes a 32x64 tile: 2 A fragments (rows mBase..+15, +16..+31)
// sharing 4 B fragments -> 8 WMMAs per K-step of 32.
// z decomposed as (b = z / Hdiv, h = z % Hdiv) with per-operand strides.
// A_F32: A source is fp32 (converted to bf16 in registers), else bf16.
// OUT_MODE: 0 = fp32 row-major (+bias,scale), 1 = bf16 row-major,
//           2 = bf16 batch-transposed: C[(row/sdim)*edim + n][sdim] layout.
template <int A_F32, int OUT_MODE>
__global__ __launch_bounds__(32) void gemm_wmma_kernel(
    const void* __restrict__ Abase, const __bf16* __restrict__ Bbase,
    const float* __restrict__ bias, void* __restrict__ Cbase,
    int K, int lda, int ldb, int ldc, float scale, int Hdiv,
    long aB, long aH, long bB, long bH, long cB, long cH,
    int sdim, int edim) {
  const int lane = threadIdx.x & 31;
  const int half = lane >> 4;
  const int l16  = lane & 15;
  const int nBase = blockIdx.x * 64;
  const int mBase = blockIdx.y * 32;
  const int z  = blockIdx.z;
  const int bz = z / Hdiv;
  const int hz = z % Hdiv;
  const long aOff = (long)bz * aB + (long)hz * aH;
  const long bOff = (long)bz * bB + (long)hz * bH;
  const long cOff = (long)bz * cB + (long)hz * cH;

  v8f acc[2][4];
#pragma unroll
  for (int u = 0; u < 2; ++u) { acc[u][0] = {}; acc[u][1] = {}; acc[u][2] = {}; acc[u][3] = {}; }

  const float*  aRowF0 = (const float*)Abase  + aOff + (long)(mBase + l16) * lda;
  const float*  aRowF1 = aRowF0 + (long)16 * lda;
  const __bf16* aRowB0 = (const __bf16*)Abase + aOff + (long)(mBase + l16) * lda;
  const __bf16* aRowB1 = aRowB0 + (long)16 * lda;
  const __bf16* bRow   = Bbase + bOff + (long)(nBase + l16) * ldb;

  for (int k0 = 0; k0 < K; k0 += 32) {
    if (k0 + 32 < K) {  // pull next K-tile toward the WGP (global_prefetch_b8)
      if (A_F32) { __builtin_prefetch(aRowF0 + k0 + 32, 0, 1); __builtin_prefetch(aRowF1 + k0 + 32, 0, 1); }
      else       { __builtin_prefetch(aRowB0 + k0 + 32, 0, 1); __builtin_prefetch(aRowB1 + k0 + 32, 0, 1); }
      __builtin_prefetch(bRow + k0 + 32, 0, 1);
    }
    v16bf a0, a1;
    if (A_F32) { a0 = load_a_f32(aRowF0 + k0, half); a1 = load_a_f32(aRowF1 + k0, half); }
    else       { a0 = load_a_bf16(aRowB0 + k0, half); a1 = load_a_bf16(aRowB1 + k0, half); }
    v16bf b0 = load_b_bf16(bRow + k0, half);
    v16bf b1 = load_b_bf16(bRow + (long)16 * ldb + k0, half);
    v16bf b2 = load_b_bf16(bRow + (long)32 * ldb + k0, half);
    v16bf b3 = load_b_bf16(bRow + (long)48 * ldb + k0, half);
    acc[0][0] = __builtin_amdgcn_wmma_f32_16x16x32_bf16(false, a0, false, b0, (short)0, acc[0][0], false, false);
    acc[0][1] = __builtin_amdgcn_wmma_f32_16x16x32_bf16(false, a0, false, b1, (short)0, acc[0][1], false, false);
    acc[0][2] = __builtin_amdgcn_wmma_f32_16x16x32_bf16(false, a0, false, b2, (short)0, acc[0][2], false, false);
    acc[0][3] = __builtin_amdgcn_wmma_f32_16x16x32_bf16(false, a0, false, b3, (short)0, acc[0][3], false, false);
    acc[1][0] = __builtin_amdgcn_wmma_f32_16x16x32_bf16(false, a1, false, b0, (short)0, acc[1][0], false, false);
    acc[1][1] = __builtin_amdgcn_wmma_f32_16x16x32_bf16(false, a1, false, b1, (short)0, acc[1][1], false, false);
    acc[1][2] = __builtin_amdgcn_wmma_f32_16x16x32_bf16(false, a1, false, b2, (short)0, acc[1][2], false, false);
    acc[1][3] = __builtin_amdgcn_wmma_f32_16x16x32_bf16(false, a1, false, b3, (short)0, acc[1][3], false, false);
  }

  // C/D layout (ISA 7.12.2): lane l16 -> column, VGPR r -> row 8*half + r
#pragma unroll
  for (int u = 0; u < 2; ++u) {
    const int mb = mBase + 16 * u;
#pragma unroll
    for (int t = 0; t < 4; ++t) {
      const int n  = nBase + 16 * t + l16;
      const float bv = bias ? bias[n] : 0.0f;
      if (OUT_MODE == 0) {
        float* C = (float*)Cbase + cOff;
#pragma unroll
        for (int r = 0; r < 8; ++r) {
          const int row = mb + 8 * half + r;
          C[(long)row * ldc + n] = acc[u][t][r] * scale + bv;
        }
      } else if (OUT_MODE == 1) {
        __bf16* C = (__bf16*)Cbase + cOff;
#pragma unroll
        for (int r = 0; r < 8; ++r) {
          const int row = mb + 8 * half + r;
          C[(long)row * ldc + n] = f2bf(acc[u][t][r] * scale + bv);
        }
      } else {  // batch-transposed bf16: [b][n][s]
        __bf16* C = (__bf16*)Cbase;
#pragma unroll
        for (int r = 0; r < 8; ++r) {
          const int row = mb + 8 * half + r;
          const int b2 = row / sdim, s2 = row % sdim;
          C[((long)b2 * edim + n) * (long)sdim + s2] = f2bf(acc[u][t][r] * scale + bv);
        }
      }
    }
  }
}

// ---------------- weight transpose + convert: W[K][N] f32 -> Wt[N][K] bf16 ----------------
__global__ __launch_bounds__(256) void transpose_bf16_kernel(
    const float* __restrict__ W, __bf16* __restrict__ Wt, int K, int N, long total) {
  long idx = (long)blockIdx.x * blockDim.x + threadIdx.x;
  if (idx >= total) return;
  long n = idx / K, k = idx % K;
  Wt[idx] = f2bf(W[k * (long)N + n]);
}

// ---------------- in-place row softmax (rowLen = 2048 fp32) ----------------
__global__ __launch_bounds__(256) void softmax_rows_kernel(float* __restrict__ data, int rowLen) {
  float* p = data + (long)blockIdx.x * rowLen;
  const int tid = threadIdx.x;
  __shared__ float red[256];
  float lmax = -3.402823466e+38f;
  for (int i = tid; i < rowLen; i += 256) lmax = fmaxf(lmax, p[i]);
  red[tid] = lmax; __syncthreads();
  for (int s = 128; s > 0; s >>= 1) { if (tid < s) red[tid] = fmaxf(red[tid], red[tid + s]); __syncthreads(); }
  const float mx = red[0]; __syncthreads();
  float lsum = 0.0f;
  for (int i = tid; i < rowLen; i += 256) { float e = __expf(p[i] - mx); p[i] = e; lsum += e; }
  red[tid] = lsum; __syncthreads();
  for (int s = 128; s > 0; s >>= 1) { if (tid < s) red[tid] += red[tid + s]; __syncthreads(); }
  const float inv = 1.0f / red[0];
  for (int i = tid; i < rowLen; i += 256) p[i] *= inv;
}

// ---------------- launch ----------------
extern "C" void kernel_launch(void* const* d_in, const int* in_sizes, int n_in,
                              void* d_out, int out_size, void* d_ws, size_t ws_size,
                              hipStream_t stream) {
  const float* query = (const float*)d_in[0];
  const float* key   = (const float*)d_in[1];
  const float* value = (const float*)d_in[2];
  const float* Wq = (const float*)d_in[3];  const float* bq = (const float*)d_in[4];
  const float* Wk = (const float*)d_in[5];  const float* bk = (const float*)d_in[6];
  const float* Wv = (const float*)d_in[7];  const float* bv = (const float*)d_in[8];
  const float* Wo = (const float*)d_in[9];  const float* bo = (const float*)d_in[10];

  float* out_attn_output  = (float*)d_out;                            // [B,S,E]
  float* out_attn_weights = (float*)d_out + (long)BB * SS * EE;       // [B,H,S,S]

  // workspace (bf16 intermediates), ~71 MB total
  char* w = (char*)d_ws;
  __bf16* wqt  = (__bf16*)w; w += (size_t)EE * QDQ * 2;   // Wq^T [E][QD]
  __bf16* wkt  = (__bf16*)w; w += (size_t)EE * KDK * 2;   // Wk^T [E][KD]
  __bf16* wvt  = (__bf16*)w; w += (size_t)EE * VDV * 2;   // Wv^T [E][VD]
  __bf16* wot  = (__bf16*)w; w += (size_t)EE * EE  * 2;   // Wo^T [E][E]
  __bf16* qb   = (__bf16*)w; w += (size_t)BB * SS * EE * 2;  // q proj [B*S][E]
  __bf16* kb   = (__bf16*)w; w += (size_t)BB * SS * EE * 2;  // k proj [B*S][E]
  __bf16* vtb  = (__bf16*)w; w += (size_t)BB * EE * SS * 2;  // v proj transposed [B][E][S]
  __bf16* ctxb = (__bf16*)w; w += (size_t)BB * SS * EE * 2;  // context [B*S][E]

  const int M = BB * SS;  // 8192

  // 1) weight transposes (f32 -> bf16 [N][K])
  {
    long t;
    t = (long)EE * QDQ; transpose_bf16_kernel<<<dim3((t + 255) / 256), 256, 0, stream>>>(Wq, wqt, QDQ, EE, t);
    t = (long)EE * KDK; transpose_bf16_kernel<<<dim3((t + 255) / 256), 256, 0, stream>>>(Wk, wkt, KDK, EE, t);
    t = (long)EE * VDV; transpose_bf16_kernel<<<dim3((t + 255) / 256), 256, 0, stream>>>(Wv, wvt, VDV, EE, t);
    t = (long)EE * EE;  transpose_bf16_kernel<<<dim3((t + 255) / 256), 256, 0, stream>>>(Wo, wot, EE, EE, t);
  }

  // 2) Q projection: qb[M,E] = query[M,QD] @ Wq + bq   (A f32, out bf16)
  gemm_wmma_kernel<1, 1><<<dim3(EE / 64, M / 32, 1), 32, 0, stream>>>(
      query, wqt, bq, qb, QDQ, QDQ, QDQ, EE, 1.0f, 1, 0, 0, 0, 0, 0, 0, 0, 0);

  // 3) K projection: kb[M,E] = key[M,KD] @ Wk + bk
  gemm_wmma_kernel<1, 1><<<dim3(EE / 64, M / 32, 1), 32, 0, stream>>>(
      key, wkt, bk, kb, KDK, KDK, KDK, EE, 1.0f, 1, 0, 0, 0, 0, 0, 0, 0, 0);

  // 4) V projection, stored transposed: vtb[b][e][s] = (value @ Wv + bv)[b,s,e]
  gemm_wmma_kernel<1, 2><<<dim3(EE / 64, M / 32, 1), 32, 0, stream>>>(
      value, wvt, bv, vtb, VDV, VDV, VDV, 0, 1.0f, 1, 0, 0, 0, 0, 0, 0, SS, EE);

  // 5) scores: attn[b,h,i,j] = (q_bh @ k_bh^T) * D^-0.5  -> fp32 into d_out
  gemm_wmma_kernel<0, 0><<<dim3(SS / 64, SS / 32, BB * HH), 32, 0, stream>>>(
      qb, kb, nullptr, out_attn_weights, DD, EE, EE, SS, 0.125f, HH,
      (long)SS * EE, (long)DD, (long)SS * EE, (long)DD,
      (long)HH * SS * SS, (long)SS * SS, 0, 0);

  // 6) softmax in place over last dim
  softmax_rows_kernel<<<dim3(BB * HH * SS), 256, 0, stream>>>(out_attn_weights, SS);

  // 7) PV: ctx[b,s,h*64+d] = P[b,h,s,:] @ v[b,:,h*64+d]   (A f32 from d_out, out bf16)
  gemm_wmma_kernel<1, 1><<<dim3(DD / 64, SS / 32, BB * HH), 32, 0, stream>>>(
      out_attn_weights, vtb, nullptr, ctxb, SS, SS, SS, EE, 1.0f, HH,
      (long)HH * SS * SS, (long)SS * SS, (long)EE * SS, (long)DD * SS,
      (long)SS * EE, (long)DD, 0, 0);

  // 8) output projection: out = ctx @ Wo + bo  -> fp32 d_out
  gemm_wmma_kernel<0, 0><<<dim3(EE / 64, M / 32, 1), 32, 0, stream>>>(
      ctxb, wot, bo, out_attn_output, EE, EE, EE, EE, 1.0f, 1, 0, 0, 0, 0, 0, 0, 0, 0);
}
